// Correlation_1151051235564
// MI455X (gfx1250) — matrix operand order repaired and checked by reading, not compiled, not run on hardware
//
#include <hip/hip_runtime.h>
#include <hip/hip_bf16.h>

typedef _Float16 v8h  __attribute__((ext_vector_type(8)));
typedef _Float16 v16h __attribute__((ext_vector_type(16)));
typedef float    v8f  __attribute__((ext_vector_type(8)));

#define C_CH   128
#define H_DIM  128
#define W_DIM  256
#define MDISP  4
#define LDSP   136   // padded channel stride in f16 (272B = 68 banks -> conflict-free b128 reads)
#define STGP   34    // staging row stride in f32 (conflict-free writes and diagonal reads)

__global__ __launch_bounds__(128)
void corr_wmma_kernel(const float* __restrict__ in1,
                      const float* __restrict__ in2,
                      float* __restrict__ out)
{
    __shared__ _Float16 sA[64][LDSP];        // [pixel][channel], f16 (pre-scaled by 1/128)
    __shared__ _Float16 sB[80][LDSP];        // [pixel][channel], f16 (x: chunk-4 .. chunk+75)
    __shared__ float    sStage[4][16][STGP]; // per-wave D staging: [wave][m][n], n in [0,32)

    const int t     = threadIdx.x;
    const int lane  = t & 31;
    const int wv    = t >> 5;        // wave 0..3, each owns a 16-px x tile
    const int half  = lane >> 4;     // 0 | 1
    const int lan15 = lane & 15;

    const int chunk = blockIdx.x * 64;
    const int y     = blockIdx.y;
    const int b     = blockIdx.z;

    const size_t plane = (size_t)H_DIM * W_DIM;
    const size_t ibase = (size_t)b * C_CH * plane;

    // ---------- load A tile: in1[b, :, y, chunk..chunk+63] -> sA[x][c]
    // (f32 -> f16, normalization 1/sumelems folded in here)
#pragma unroll 4
    for (int it = 0; it < 16; ++it) {
        int idx = it * 128 + t;              // < 2048 float4s
        int c   = idx >> 4;
        int xq  = (idx & 15) << 2;
        const float4 v = *(const float4*)(in1 + ibase + (size_t)c * plane
                                          + (size_t)y * W_DIM + chunk + xq);
        sA[xq + 0][c] = (_Float16)(v.x * (1.0f / 128.0f));
        sA[xq + 1][c] = (_Float16)(v.y * (1.0f / 128.0f));
        sA[xq + 2][c] = (_Float16)(v.z * (1.0f / 128.0f));
        sA[xq + 3][c] = (_Float16)(v.w * (1.0f / 128.0f));
    }
    __syncthreads();

    // ---------- hoisted A fragments (invariant over dy); contiguous k order: element s of half h holds k = 16h + s
    const int apix = wv * 16 + lan15;
    v16h afrag[4];
#pragma unroll
    for (int ks = 0; ks < 4; ++ks) {
        const int kb = ks * 32 + (half ? 16 : 0);
        v8h lo = *(const v8h*)&sA[apix][kb];        // K = kb .. kb+7
        v8h hi = *(const v8h*)&sA[apix][kb + 8];    // K = kb+8 .. kb+15
        afrag[ks] = __builtin_shufflevector(lo, hi, 0,1,2,3,4,5,6,7,8,9,10,11,12,13,14,15);
    }

    const int x0 = chunk + wv * 16;   // this wave's output-tile base x
    const int j0 = wv * 16;           // B tile0 start in sB; tile1 = j0+16

    // output base for this wave's tile (channel stride = plane)
    const size_t obase = ((size_t)b * 81) * plane + (size_t)y * W_DIM + x0;

    for (int dy = -MDISP; dy <= MDISP; ++dy) {
        __syncthreads();              // previous iter's sB readers done
        // ------ load B tile: in2[b, :, y+dy, chunk-4 .. chunk+75] (zero-pad OOB)
        const int  gy    = y + dy;
        const bool rowok = (gy >= 0) && (gy < H_DIM);
#pragma unroll 4
        for (int it = 0; it < 20; ++it) {
            int idx = it * 128 + t;           // < 2560 float4s
            int c   = idx / 20;
            int jj  = (idx - c * 20) << 2;
            int gx  = chunk - 4 + jj;         // gx % 4 == 0 -> float4 fully in or fully out
            float4 v = make_float4(0.f, 0.f, 0.f, 0.f);
            if (rowok && gx >= 0 && gx + 3 < W_DIM) {
                const float* p = in2 + ibase + (size_t)c * plane + (size_t)gy * W_DIM + gx;
                v = *(const float4*)p;
                if (gy + 1 < H_DIM) __builtin_prefetch(p + W_DIM, 0, 1); // next dy row
            }
            sB[jj + 0][c] = (_Float16)v.x;
            sB[jj + 1][c] = (_Float16)v.y;
            sB[jj + 2][c] = (_Float16)v.z;
            sB[jj + 3][c] = (_Float16)v.w;
        }
        __syncthreads();

        // ------ bulk-load all 8 B fragments, then issue the 8 WMMAs back-to-back
        v16h bf0[4], bf1[4];
#pragma unroll
        for (int ks = 0; ks < 4; ++ks) {
            const int kb = ks * 32 + (half ? 16 : 0);   // contiguous k = 16*half + s, same order as A
            v8h b0lo = *(const v8h*)&sB[j0 + lan15][kb];
            v8h b0hi = *(const v8h*)&sB[j0 + lan15][kb + 8];
            bf0[ks] = __builtin_shufflevector(b0lo, b0hi, 0,1,2,3,4,5,6,7,8,9,10,11,12,13,14,15);
            v8h b1lo = *(const v8h*)&sB[j0 + 16 + lan15][kb];
            v8h b1hi = *(const v8h*)&sB[j0 + 16 + lan15][kb + 8];
            bf1[ks] = __builtin_shufflevector(b1lo, b1hi, 0,1,2,3,4,5,6,7,8,9,10,11,12,13,14,15);
        }
        __builtin_amdgcn_sched_barrier(0);   // keep all ds_loads ahead of the WMMA chain

        v8f acc0 = {};
        v8f acc1 = {};
#pragma unroll
        for (int ks = 0; ks < 4; ++ks) {
            acc0 = __builtin_amdgcn_wmma_f32_16x16x32_f16(false, afrag[ks], false, bf0[ks],
                                                          (short)0, acc0, false, false);
            acc1 = __builtin_amdgcn_wmma_f32_16x16x32_f16(false, afrag[ks], false, bf1[ks],
                                                          (short)0, acc1, false, false);
        }
        __builtin_amdgcn_sched_barrier(0);   // don't let stores sink into the WMMA chain

        // ------ stage both accumulators into LDS (branch-free), then coalesced stores.
        // stage[m][n]: n<16 -> D0 col n (dx = n-m-4), n>=16 -> D1 col n-16 (dx = n-m+12).
        // Output (m, d) for d in [0,9) is stage[m][m+d] in both cases.
#pragma unroll
        for (int r = 0; r < 8; ++r) {
            const int M = r + half * 8;      // ISA f32 C/D 16x16 layout
            sStage[wv][M][lan15]      = acc0[r];
            sStage[wv][M][lan15 + 16] = acc1[r];
        }

        const int dyc = (dy + MDISP) * 9;
#pragma unroll
        for (int i = 0; i < 4; ++i) {        // 128 of 144 values: fully uniform
            const int idx = i * 32 + lane;
            const int d   = idx >> 4;
            const int m   = idx & 15;
            out[obase + (size_t)(dyc + d) * plane + m] = sStage[wv][m][m + d];
        }
        {                                     // last 16 values (lanes 0-15 only)
            const int idx = 128 + lane;
            if (idx < 144) {
                const int d = idx >> 4;       // == 8
                const int m = idx & 15;
                out[obase + (size_t)(dyc + d) * plane + m] = sStage[wv][m][m + d];
            }
        }
    }
}

extern "C" void kernel_launch(void* const* d_in, const int* in_sizes, int n_in,
                              void* d_out, int out_size, void* d_ws, size_t ws_size,
                              hipStream_t stream) {
    const float* in1 = (const float*)d_in[0];
    const float* in2 = (const float*)d_in[1];
    float* out = (float*)d_out;
    (void)in_sizes; (void)n_in; (void)out_size; (void)d_ws; (void)ws_size;

    dim3 grid(W_DIM / 64, H_DIM, 8);   // (x-chunks, rows, batch)
    dim3 block(128);                   // 4 waves, one 16-px tile each
    corr_wmma_kernel<<<grid, block, 0, stream>>>(in1, in2, out);
}
